// Model_10711648436736
// MI455X (gfx1250) — matrix operand order repaired and checked by reading, not compiled, not run on hardware
//
#include <hip/hip_runtime.h>
#include <hip/hip_bf16.h>
#include <math.h>

// ---------------------------------------------------------------------------
// Model constants
// ---------------------------------------------------------------------------
#define BATCH   16
#define TSTEPS  3300
#define FEAT    80
#define NCLS    512
#define RNN     512
#define DIN     592          // NCLS + FEAT
#define G3      1536         // 3*RNN
#define NWG     16           // workgroups cooperating on the GRU scan
#define CROP    550          // INDENT
#define L3      4400         // stage-3 upsample length before crop

typedef __attribute__((ext_vector_type(16))) __bf16 v16bf;
typedef __attribute__((ext_vector_type(8)))  __bf16 bf16x8;
typedef __attribute__((ext_vector_type(8)))  float  v8f;
typedef __attribute__((ext_vector_type(4)))  unsigned int u32x4;
typedef __attribute__((ext_vector_type(8)))  int    i32x8;
typedef __attribute__((ext_vector_type(4)))  int    i32x4;

// ---------------------------------------------------------------------------
// WMMA fragment helpers (wave32; A/B use this code's contiguous k order, not the ISA table order)
//   A 16x32 bf16: lane -> M=lane%16, elems = 16 consecutive K at k0+hi*16
//   B 32x16 bf16: lane -> N=lane%16, elems = 16 consecutive K at k0+hi*16
//   C 16x16 f32 : lane -> N=lane%16, VGPR v -> M = v + hi*8
// ---------------------------------------------------------------------------
__device__ __forceinline__ v16bf a_frag(const __bf16* base, int rowStride,
                                        int m, int k0, int hi) {
  const __bf16* p = base + m * rowStride + k0 + hi * 16;
  bf16x8 lo = *(const bf16x8*)(p);
  bf16x8 hh = *(const bf16x8*)(p + 8);
  return __builtin_shufflevector(lo, hh, 0,1,2,3,4,5,6,7,8,9,10,11,12,13,14,15);
}

__device__ __forceinline__ v8f wmma_bf16(v16bf a, v16bf b, v8f c) {
  return __builtin_amdgcn_wmma_f32_16x16x32_bf16(false, a, false, b,
                                                 (short)0, c, false, false);
}

// ---------------------------------------------------------------------------
// Tensor Data Mover descriptors (CDNA5 ISA)
// Generic LDS address: low 32 bits of a generic pointer to __shared__ memory
// are the LDS byte address (flat aperture mapping truncates upper bits).
// This toolchain exposes the 6-arg builtin:
//   (u32x4 g0, i32x8 g1, i32x4 g2, i32x4 g3, i32x8 extra, i32 cpol)
// ---------------------------------------------------------------------------
__device__ __forceinline__ unsigned int lds_off(const void* p) {
  return (unsigned int)(uintptr_t)p;
}

// 2D tile load: tile_h rows of tile_w 2-byte elements; row stride in elements.
// tile_h == 1 / stride == tile_w gives a plain 1D contiguous copy.
__device__ __forceinline__ void tdm_load_2d(unsigned int ldsAddr, const void* gptr,
                                            unsigned int tile_w, unsigned int tile_h,
                                            unsigned long long strideElems) {
  unsigned long long ga = (unsigned long long)(uintptr_t)gptr;
  u32x4 g0;
  g0[0] = 1u;                                         // count=1, user mode
  g0[1] = ldsAddr;                                    // lds_addr (bytes)
  g0[2] = (unsigned int)ga;                           // global_addr[31:0]
  g0[3] = (unsigned int)((ga >> 32) & 0x1FFFFFFu)     // global_addr[56:32]
        | (2u << 30);                                 // type = 2 ("image")
  i32x8 g1;
  unsigned int td0 = tile_w;                          // tensor_dim0 (elements)
  unsigned int td1 = tile_h;                          // tensor_dim1
  unsigned int s0lo = (unsigned int)(strideElems & 0xFFFFFFFFull);
  unsigned int s0hi = (unsigned int)((strideElems >> 32) & 0xFFFFull);
  g1[0] = (int)(1u << 16);                            // data_size=1 (2 bytes)
  g1[1] = (int)((td0 & 0xFFFFu) << 16);               // tensor_dim0[15:0]
  g1[2] = (int)(((td0 >> 16) & 0xFFFFu) | ((td1 & 0xFFFFu) << 16));
  g1[3] = (int)(((td1 >> 16) & 0xFFFFu) | ((tile_w & 0xFFFFu) << 16)); // tile_dim0
  g1[4] = (int)(tile_h > 1 ? (tile_h & 0xFFFFu) : 0u);// tile_dim1 (0=unused), tile_dim2=0
  g1[5] = (int)s0lo;                                  // tensor_dim0_stride[31:0]
  g1[6] = (int)s0hi;                                  // stride0[47:32], stride1[15:0]=0
  g1[7] = 0;                                          // stride1[47:16]=0
  i32x4 z4 = {0, 0, 0, 0};
  i32x8 z8 = {0, 0, 0, 0, 0, 0, 0, 0};
  __builtin_amdgcn_tensor_load_to_lds(g0, g1, z4, z4, z8, 0);
}

// ---------------------------------------------------------------------------
// prep: f32 -> bf16 weight conversion + state/counter init (every launch)
// ---------------------------------------------------------------------------
__global__ void prep_kernel(const float* __restrict__ w_ih,
                            const float* __restrict__ w_hh,
                            const float* __restrict__ fc1_w,
                            const float* __restrict__ fc2_w,
                            __bf16* __restrict__ whh_b,
                            __bf16* __restrict__ wih_b,
                            __bf16* __restrict__ fc1b,
                            __bf16* __restrict__ fc2b,
                            __bf16* __restrict__ hbuf,
                            unsigned* __restrict__ ctr) {
  const size_t i0 = (size_t)blockIdx.x * blockDim.x + threadIdx.x;
  const size_t st = (size_t)gridDim.x * blockDim.x;
  for (size_t i = i0; i < (size_t)G3 * RNN; i += st) whh_b[i] = (__bf16)w_hh[i];
  for (size_t i = i0; i < (size_t)G3 * FEAT; i += st) {
    size_t g = i / FEAT, f = i % FEAT;
    wih_b[i] = (__bf16)w_ih[g * DIN + NCLS + f];
  }
  for (size_t i = i0; i < (size_t)NCLS * RNN; i += st) {
    fc1b[i] = (__bf16)fc1_w[i];
    fc2b[i] = (__bf16)fc2_w[i];
  }
  for (size_t i = i0; i < (size_t)2 * BATCH * RNN; i += st) hbuf[i] = (__bf16)0.0f;
  if (i0 == 0) *ctr = 0u;
}

// ---------------------------------------------------------------------------
// Mel upsample stage: repeat by sc then box-filter k=2sc+1, pad sc each side.
// out[b,f,t] = (1/k) * sum_{d=-sc..sc} in[b,f,(t+d)/sc]  (zero outside)
// ---------------------------------------------------------------------------
__global__ void upsample_stage(const float* __restrict__ in,
                               float* __restrict__ out,
                               int Lin, int sc, int total) {
  int idx = blockIdx.x * blockDim.x + threadIdx.x;
  if (idx >= total) return;
  const int Lout = Lin * sc;
  const int k = 2 * sc + 1;
  int t = idx % Lout;
  int bf = idx / Lout;
  const float* ip = in + (size_t)bf * Lin;
  float s = 0.f;
  for (int d = -sc; d <= sc; ++d) {
    int u = t + d;
    if (u >= 0 && u < Lout) s += ip[u / sc];
  }
  out[idx] = s / (float)k;
}

// ---------------------------------------------------------------------------
// Persistent GRU scan: 16 WGs x 192 threads (6 waves). WG wg owns hidden
// units [wg*32, wg*32+32) -> its 96 gate rows of w_hh pinned in LDS via one
// TDM 2D tile load. Per step: TDM stages the 16KB hidden state, 6 waves x
// (16 recurrent + 3 mel-proj) WMMAs, one-hot gather into the C fragment,
// gating in VALU, ping-pong global h, L2-atomic step barrier.
// ---------------------------------------------------------------------------
__global__ __launch_bounds__(192) void gru_kernel(
    const int*    __restrict__ x,      // (B,T)
    const float*  __restrict__ mel3,   // (B,FEAT,L3) stage-3 output
    const float*  __restrict__ w_ih,   // (G3,DIN) f32 (for one-hot gather)
    const float*  __restrict__ b_ih,
    const float*  __restrict__ b_hh,
    const __bf16* __restrict__ whh_b,  // (G3,RNN) bf16
    const __bf16* __restrict__ wih_b,  // (G3,FEAT) bf16
    __bf16*       __restrict__ hbuf,   // 2 x (B,RNN) ping-pong
    __bf16*       __restrict__ hs,     // (B*T,RNN)
    unsigned*     __restrict__ ctr) {
  __shared__ __align__(64) __bf16 whh_l[96 * RNN];   // 96 KB
  __shared__ __align__(64) __bf16 wih_l[96 * 96];    // 18 KB (K padded 80->96)
  __shared__ __align__(64) __bf16 h_l[BATCH * RNN];  // 16 KB
  __shared__ __align__(64) __bf16 mel_l[BATCH * 96]; //  3 KB
  __shared__ __align__(64) float  gate_l[8 * 256];   //  8 KB
  __shared__ float bih_l[96];
  __shared__ float bhh_l[96];
  __shared__ int   xd_l[BATCH];

  const int tid = threadIdx.x;
  const int wg  = blockIdx.x;            // 0..15
  const int j0  = wg * 32;               // first owned hidden unit
  const int wv  = tid >> 5;              // wave 0..5
  const int L   = tid & 31;
  const int hi  = L >> 4;
  const int ln  = L & 15;
  const int cLocal = wv * 16 + ln;       // local gate column 0..95
  const int gRow   = (wv >> 1) * RNN + j0 + ((wv & 1) * 16 + ln); // global gate row

  // -- pin this WG's w_hh slice in LDS: one TDM 2D tile (3 rows x 32KB) -----
  if (wv == 0) {
    tdm_load_2d(lds_off(whh_l), whh_b + (size_t)j0 * RNN,
                32u * RNN /*tile_w elems*/, 3u /*tile_h*/,
                (unsigned long long)RNN * RNN /*row stride elems*/);
  }
  for (int idx = tid; idx < 96 * 96; idx += 192) {
    int c = idx / 96, f = idx % 96;
    int gg = (c >> 5) * RNN + j0 + (c & 31);
    wih_l[idx] = (f < FEAT) ? wih_b[(size_t)gg * FEAT + f] : (__bf16)0.0f;
  }
  for (int idx = tid; idx < 96; idx += 192) {
    int gg = (idx >> 5) * RNN + j0 + (idx & 31);
    bih_l[idx] = b_ih[gg];
    bhh_l[idx] = b_hh[gg];
  }
  if (wv == 0) __builtin_amdgcn_s_wait_tensorcnt(0);
  __syncthreads();

  const __bf16* wrow  = whh_l + cLocal * RNN;
  const __bf16* wxrow = wih_l + cLocal * 96;
  const float*  wgrow = w_ih + (size_t)gRow * DIN;   // one-hot gather row

  for (int t = 0; t < TSTEPS; ++t) {
    const __bf16* hcur = hbuf + (size_t)(t & 1) * (BATCH * RNN);
    __bf16*       hnxt = hbuf + (size_t)((t + 1) & 1) * (BATCH * RNN);

    // stage hidden state via TDM (one instruction, 16KB contiguous)
    if (wv == 0) {
      tdm_load_2d(lds_off(h_l), hcur, (unsigned)(BATCH * RNN), 1u,
                  (unsigned long long)(BATCH * RNN));
    }
    // stage mel frame (transpose + pad + cvt) and sample indices
    for (int idx = tid; idx < BATCH * 96; idx += 192) {
      int m = idx / 96, f = idx % 96;
      mel_l[idx] = (f < FEAT)
          ? (__bf16)mel3[((size_t)m * FEAT + f) * L3 + CROP + t]
          : (__bf16)0.0f;
    }
    if (tid < BATCH) xd_l[tid] = x[(size_t)tid * TSTEPS + t];
    if (wv == 0) __builtin_amdgcn_s_wait_tensorcnt(0);
    __syncthreads();

    // recurrent: hp = h @ w_hh^T  (K=512 -> 16 WMMAs)
    v8f accH = {};
    #pragma unroll
    for (int kt = 0; kt < 16; ++kt) {
      v16bf a = a_frag(h_l, RNN, ln, kt * 32, hi);
      v16bf b = *(const v16bf*)(wrow + kt * 32 + hi * 16);
      accH = wmma_bf16(a, b, accH);
    }
    // input proj (mel part): K=96 -> 3 WMMAs
    v8f accX = {};
    #pragma unroll
    for (int kt = 0; kt < 3; ++kt) {
      v16bf a = a_frag(mel_l, 96, ln, kt * 32, hi);
      v16bf b = *(const v16bf*)(wxrow + kt * 32 + hi * 16);
      accX = wmma_bf16(a, b, accX);
    }
    // one-hot gather + input bias directly into the C fragment
    const float bih = bih_l[cLocal];
    const float bhh = bhh_l[cLocal];
    #pragma unroll
    for (int v = 0; v < 8; ++v) {
      int m = v + hi * 8;                       // batch row of this element
      accX[v] += bih + wgrow[xd_l[m]];
    }
    // stage gate tiles: r,z combined; n keeps x/h separate (tanh(xn + r*hn))
    if (wv < 4) {
      #pragma unroll
      for (int v = 0; v < 8; ++v)
        gate_l[wv * 256 + (v + hi * 8) * 16 + ln] = accH[v] + accX[v] + bhh;
    } else {
      #pragma unroll
      for (int v = 0; v < 8; ++v) {
        gate_l[wv * 256 + (v + hi * 8) * 16 + ln]       = accH[v] + bhh; // slots 4,5
        gate_l[(wv + 2) * 256 + (v + hi * 8) * 16 + ln] = accX[v];        // slots 6,7
      }
    }
    __syncthreads();

    // gating for the 16x32 owned block of h_new
    for (int idx = tid; idx < BATCH * 32; idx += 192) {
      int m = idx >> 5, nl = idx & 31;
      int sl = nl >> 4, nn = nl & 15;
      float r  = gate_l[(0 + sl) * 256 + m * 16 + nn];
      float z  = gate_l[(2 + sl) * 256 + m * 16 + nn];
      float hn = gate_l[(4 + sl) * 256 + m * 16 + nn];
      float xn = gate_l[(6 + sl) * 256 + m * 16 + nn];
      r = 1.f / (1.f + __expf(-r));
      z = 1.f / (1.f + __expf(-z));
      float nv = tanhf(xn + r * hn);
      float hp = (float)h_l[m * RNN + j0 + nl];
      float hnew = (1.f - z) * nv + z * hp;
      hnxt[m * RNN + j0 + nl] = (__bf16)hnew;
      hs[((size_t)m * TSTEPS + t) * RNN + j0 + nl] = (__bf16)hnew;
    }

    // cross-workgroup step barrier (single L2 atomic + sleep-spin)
    __threadfence();
    __syncthreads();
    if (tid == 0) {
      atomicAdd(ctr, 1u);
      const unsigned tgt = (unsigned)(t + 1) * NWG;
      while (__hip_atomic_load(ctr, __ATOMIC_ACQUIRE, __HIP_MEMORY_SCOPE_AGENT) < tgt)
        __builtin_amdgcn_s_sleep(2);
    }
    __syncthreads();
  }
}

// ---------------------------------------------------------------------------
// Fused fc1(relu) + fc2 over 16-row tiles of hs: TDM stages the activation
// tile; 8 waves x 4 n-tiles each, K=512 (16 WMMAs) per stage; weight tiles
// prefetched (global_prefetch_b8).
// ---------------------------------------------------------------------------
__global__ __launch_bounds__(256) void fc_kernel(
    const __bf16* __restrict__ hs,       // (B*T,RNN) bf16
    const __bf16* __restrict__ fc1b,     // (NCLS,RNN) bf16
    const float*  __restrict__ fc1_bias,
    const __bf16* __restrict__ fc2b,     // (NCLS,NCLS) bf16
    const float*  __restrict__ fc2_bias,
    float*        __restrict__ out) {    // (B*T,NCLS) f32
  __shared__ __align__(64) __bf16 a_l[16 * RNN];
  __shared__ __align__(64) __bf16 a1_l[16 * NCLS];

  const int tid = threadIdx.x;
  const int wv = tid >> 5, L = tid & 31, hi = L >> 4, ln = L & 15;
  const size_t r0 = (size_t)blockIdx.x * 16;

  if (wv == 0) {
    tdm_load_2d(lds_off(a_l), hs + r0 * RNN, (unsigned)(16 * RNN), 1u,
                (unsigned long long)(16 * RNN));
    __builtin_amdgcn_s_wait_tensorcnt(0);
  }
  __syncthreads();

  // fc1 + relu -> a1_l (bf16)
  #pragma unroll
  for (int q = 0; q < 4; ++q) {
    const int nt = wv + q * 8;
    const int n = nt * 16 + ln;
    const __bf16* wrow = fc1b + (size_t)n * RNN;
    v8f acc = {};
    #pragma unroll
    for (int kt = 0; kt < 16; ++kt) {
      __builtin_prefetch(wrow + kt * 32 + 256, 0, 1);
      v16bf a = a_frag(a_l, RNN, ln, kt * 32, hi);
      v16bf b = *(const v16bf*)(wrow + kt * 32 + hi * 16);
      acc = wmma_bf16(a, b, acc);
    }
    const float bias = fc1_bias[n];
    #pragma unroll
    for (int v = 0; v < 8; ++v) {
      int m = v + hi * 8;
      a1_l[m * NCLS + n] = (__bf16)fmaxf(acc[v] + bias, 0.f);
    }
  }
  __syncthreads();

  // fc2 -> out (f32)
  #pragma unroll
  for (int q = 0; q < 4; ++q) {
    const int nt = wv + q * 8;
    const int n = nt * 16 + ln;
    const __bf16* wrow = fc2b + (size_t)n * NCLS;
    v8f acc = {};
    #pragma unroll
    for (int kt = 0; kt < 16; ++kt) {
      __builtin_prefetch(wrow + kt * 32 + 256, 0, 1);
      v16bf a = a_frag(a1_l, NCLS, ln, kt * 32, hi);
      v16bf b = *(const v16bf*)(wrow + kt * 32 + hi * 16);
      acc = wmma_bf16(a, b, acc);
    }
    const float bias = fc2_bias[n];
    #pragma unroll
    for (int v = 0; v < 8; ++v) {
      int m = v + hi * 8;
      out[(r0 + m) * NCLS + n] = acc[v] + bias;
    }
  }
}

// ---------------------------------------------------------------------------
// Host launcher
// ---------------------------------------------------------------------------
extern "C" void kernel_launch(void* const* d_in, const int* in_sizes, int n_in,
                              void* d_out, int out_size, void* d_ws, size_t ws_size,
                              hipStream_t stream) {
  const int*   x      = (const int*)  d_in[0];
  const float* mels   = (const float*)d_in[1];
  // d_in[2..4]: up_k0..2 (uniform 1/k box kernels, values folded into kernels)
  const float* w_ih   = (const float*)d_in[5];
  const float* w_hh   = (const float*)d_in[6];
  const float* b_ih   = (const float*)d_in[7];
  const float* b_hh   = (const float*)d_in[8];
  const float* fc1_w  = (const float*)d_in[9];
  const float* fc1_bv = (const float*)d_in[10];
  const float* fc2_w  = (const float*)d_in[11];
  const float* fc2_bv = (const float*)d_in[12];
  float* out = (float*)d_out;

  char* ws = (char*)d_ws;
  auto alloc = [&](size_t bytes) -> char* {
    char* p = ws;
    ws += (bytes + 255) & ~(size_t)255;
    return p;
  };
  float*    s1    = (float*)   alloc((size_t)BATCH * FEAT * 80   * 4);
  float*    s2    = (float*)   alloc((size_t)BATCH * FEAT * 400  * 4);
  float*    s3    = (float*)   alloc((size_t)BATCH * FEAT * L3   * 4);
  __bf16*   whh_b = (__bf16*)  alloc((size_t)G3 * RNN * 2);
  __bf16*   wih_b = (__bf16*)  alloc((size_t)G3 * FEAT * 2);
  __bf16*   fc1b  = (__bf16*)  alloc((size_t)NCLS * RNN * 2);
  __bf16*   fc2b  = (__bf16*)  alloc((size_t)NCLS * NCLS * 2);
  __bf16*   hbuf  = (__bf16*)  alloc((size_t)2 * BATCH * RNN * 2);
  __bf16*   hsb   = (__bf16*)  alloc((size_t)BATCH * TSTEPS * RNN * 2);
  unsigned* ctr   = (unsigned*)alloc(256);

  prep_kernel<<<512, 256, 0, stream>>>(w_ih, w_hh, fc1_w, fc2_w,
                                       whh_b, wih_b, fc1b, fc2b, hbuf, ctr);

  const int tot0 = BATCH * FEAT * 80;
  const int tot1 = BATCH * FEAT * 400;
  const int tot2 = BATCH * FEAT * L3;
  upsample_stage<<<(tot0 + 255) / 256, 256, 0, stream>>>(mels, s1, 16, 5, tot0);
  upsample_stage<<<(tot1 + 255) / 256, 256, 0, stream>>>(s1, s2, 80, 5, tot1);
  upsample_stage<<<(tot2 + 255) / 256, 256, 0, stream>>>(s2, s3, 400, 11, tot2);

  gru_kernel<<<NWG, 192, 0, stream>>>(x, s3, w_ih, b_ih, b_hh,
                                      whh_b, wih_b, hbuf, hsb, ctr);

  fc_kernel<<<(BATCH * TSTEPS) / 16, 256, 0, stream>>>(hsb, fc1b, fc1_bv,
                                                       fc2b, fc2_bv, out);
}
